// IntraDiffCov_89103391523710
// MI455X (gfx1250) — matrix operand order repaired and checked by reading, not compiled, not run on hardware
//
#include <hip/hip_runtime.h>
#include <stdint.h>

#define V_   3
#define N_   4096
#define IN_  256
#define OUT_ 128
#define NT_  (N_/16)          // 256 row-tiles per view
#define NWAVE_ (V_*NT_)       // 768 tile-waves

typedef uint32_t u32;
typedef uint16_t u16;
typedef unsigned long long u64;
typedef __attribute__((ext_vector_type(16))) __bf16 v16bf;
typedef __attribute__((ext_vector_type(8)))  float  v8f;
typedef __attribute__((ext_vector_type(8)))  u32    v8u;
typedef __attribute__((ext_vector_type(4)))  u32    v4u;

struct v4u2 { v4u a, b; };

// 16 contiguous bf16 (32B) -> one WMMA operand fragment (per-lane view)
static __device__ __forceinline__ v16bf ld32B(const u16* p) {
  return __builtin_bit_cast(v16bf, *(const v8u*)p);
}
// two disjoint 16B pieces (A operand, contiguous k: 16h..16h+7, then 16h+8..16h+15)
static __device__ __forceinline__ v16bf ld16Bx2(const u16* p0, const u16* p1) {
  v4u2 t; t.a = *(const v4u*)p0; t.b = *(const v4u*)p1;
  return __builtin_bit_cast(v16bf, t);
}
static __device__ __forceinline__ u32 bf16_1(float x) {        // f32 -> bf16 (RNE)
  u32 a = __builtin_bit_cast(u32, x);
  a += 0x7fffu + ((a >> 16) & 1u);
  return a >> 16;
}
static __device__ __forceinline__ u32 pkbf(float lo, float hi) { // pack 2x bf16
  u32 a = __builtin_bit_cast(u32, lo);
  u32 b = __builtin_bit_cast(u32, hi);
  a += 0x7fffu + ((a >> 16) & 1u);
  b += 0x7fffu + ((b >> 16) & 1u);
  return (a >> 16) | (b & 0xffff0000u);
}
static __device__ __forceinline__ float sigmoidf_(float x) {
  return 1.0f / (1.0f + __expf(-x));
}
static __device__ __forceinline__ v8f vzero() {
  v8f z;
#pragma unroll
  for (int r = 0; r < 8; ++r) z[r] = 0.f;
  return z;
}
#define WMMA_BF16(A,B,C) \
  __builtin_amdgcn_wmma_f32_16x16x32_bf16(false,(A),false,(B),(short)0,(C),false,false)

// Async global->LDS copy, 16B per lane, tracked by ASYNCcnt (CDNA5).
static __device__ __forceinline__ void async_g2l_b128(u32 lds_off, u64 gaddr) {
  asm volatile("global_load_async_to_lds_b128 %0, %1, off"
               :: "v"(lds_off), "v"(gaddr) : "memory");
}
static __device__ __forceinline__ void wait_async0() {
  asm volatile("s_wait_asynccnt 0x0" ::: "memory");
}
static __device__ __forceinline__ u32 lds_off_of(const void* p) {
  return (u32)(uintptr_t)p;   // flat addr low 32 bits == wave-relative LDS addr
}

// ---------------------------------------------------------------------------
// f32 -> bf16 bulk conversion
__global__ void k_cvt_bf16(const float* __restrict__ src, u16* __restrict__ dst, int n) {
  int i = blockIdx.x * blockDim.x + threadIdx.x;
  if (i < n) dst[i] = (u16)bf16_1(src[i]);
}

// ---------------------------------------------------------------------------
// One projection matrix for one 16-row n-tile: Of[n0..n0+15][0..127] = X W^T + b.
static __device__ __forceinline__ void proj_mat(const v16bf A[8],
                                                const u16* __restrict__ Wb,
                                                const float* __restrict__ bias,
                                                float* __restrict__ Of,
                                                int n0, int lm, bool hi)
{
#pragma unroll
  for (int t = 0; t < 8; ++t) {
    const int o = t*16 + lm;
    const float bval = bias[o];
    v8f acc;
#pragma unroll
    for (int r = 0; r < 8; ++r) acc[r] = bval;
    const u16* wrow = Wb + (size_t)o * IN_;
    v16bf Bcur = ld32B(wrow + (hi ? 16 : 0));
#pragma unroll
    for (int c = 0; c < 8; ++c) {
      v16bf Bnext = Bcur;
      if (c < 7) Bnext = ld32B(wrow + (c+1)*32 + (hi ? 16 : 0));  // issue before WMMA
      acc = WMMA_BF16(A[c], Bcur, acc);
      Bcur = Bnext;
    }
#pragma unroll
    for (int r = 0; r < 8; ++r) {
      const int n = n0 + r + (hi ? 8 : 0);
      Of[(size_t)n * OUT_ + o] = acc[r];
    }
  }
}

// Q/K/V projections: one wave per 16-row n-tile per view.
__global__ __launch_bounds__(32)
void k_proj(const u16* __restrict__ Xb,
            const u16* __restrict__ Wqb, const u16* __restrict__ Wkb,
            const u16* __restrict__ Wvb,
            const float* __restrict__ bq, const float* __restrict__ bk,
            const float* __restrict__ bv,
            float* __restrict__ Qf, float* __restrict__ Kf, float* __restrict__ Vf)
{
  const int wid = blockIdx.x;
  const int v   = wid / NT_;
  const int n0  = (wid % NT_) * 16;
  const int lane = threadIdx.x & 31;
  const int lm   = lane & 15;
  const bool hi  = lane >= 16;

  v16bf A[8];
  const u16* xrow = Xb + ((size_t)(v*N_ + n0 + lm)) * IN_;
#pragma unroll
  for (int c = 0; c < 8; ++c) {
    int e = c*32 + (hi ? 16 : 0);
    A[c] = ld16Bx2(xrow + e, xrow + e + 8);
  }

  const size_t wo = (size_t)v * OUT_ * IN_;
  const size_t oo = (size_t)(v*N_) * OUT_;
  proj_mat(A, Wqb + wo, bq + v*OUT_, Qf + oo, n0, lm, hi);
  proj_mat(A, Wkb + wo, bk + v*OUT_, Kf + oo, n0, lm, hi);
  proj_mat(A, Wvb + wo, bv + v*OUT_, Vf + oo, n0, lm, hi);
}

// ---------------------------------------------------------------------------
// Deterministic two-stage Frobenius sum-of-squares.
__global__ void k_sumsq(const float* __restrict__ Qf, const float* __restrict__ Kf,
                        float* __restrict__ part)
{
  const int b  = blockIdx.x;                 // 0..383
  const float* src = (b < 192) ? Qf : Kf;
  const int rb = b % 192;
  const int v  = rb / 64, k = rb % 64;
  const float* p = src + (size_t)v * (N_*OUT_) + (size_t)k * 8192;
  const int t = threadIdx.x;                 // 256 threads
  float s = 0.f;
  for (int i = 0; i < 32; ++i) { float x = p[t + i*256]; s += x*x; }
  __shared__ float red[256];
  red[t] = s; __syncthreads();
  for (int off = 128; off > 0; off >>= 1) {
    if (t < off) red[t] += red[t + off];
    __syncthreads();
  }
  if (t == 0) part[b] = red[0];
}

__global__ void k_norm_reduce(const float* __restrict__ part, float* __restrict__ inv) {
  int t = threadIdx.x;
  if (t < 6) {
    int v = t % 3;
    int base = (t < 3 ? 0 : 192) + v*64;
    float s = 0.f;
    for (int i = 0; i < 64; ++i) s += part[base + i];
    inv[t] = rsqrtf(s);
  }
}

// ---------------------------------------------------------------------------
__global__ void k_normalize(const float* __restrict__ Qf, const float* __restrict__ Kf,
                            const float* __restrict__ inv,
                            u16* __restrict__ qb, u16* __restrict__ kb) {
  int i = blockIdx.x * blockDim.x + threadIdx.x;
  if (i >= V_*N_*OUT_) return;
  int v = i >> 19;                       // N_*OUT_ = 2^19
  qb[i] = (u16)bf16_1(Qf[i] * inv[v]);
  kb[i] = (u16)bf16_1(Kf[i] * inv[3+v]);
}

// ---------------------------------------------------------------------------
// rowsum[j] = sum_m sigmoid(qn_j . kn_m).  One wave per 16-row j-tile,
// double-buffered B fragments (two m-tiles per loop body).
__global__ __launch_bounds__(32)
void k_rowsum(const u16* __restrict__ qb, const u16* __restrict__ kb,
              float* __restrict__ rowsum)
{
  const int wid = blockIdx.x;
  const int v   = wid / NT_;
  const int j0  = (wid % NT_) * 16;
  const int lane = threadIdx.x & 31;
  const int lm   = lane & 15;
  const bool hi  = lane >= 16;

  v16bf Aq[4];
  const u16* qrow = qb + ((size_t)(v*N_ + j0 + lm)) * OUT_;
#pragma unroll
  for (int c = 0; c < 4; ++c) {
    int e = c*32 + (hi ? 16 : 0);
    Aq[c] = ld16Bx2(qrow + e, qrow + e + 8);
  }

  float sacc[8];
#pragma unroll
  for (int r = 0; r < 8; ++r) sacc[r] = 0.f;

  const u16* kbase = kb + (size_t)(v*N_) * OUT_;
  const int boff = hi ? 16 : 0;

  v16bf B0[4], B1[4];
  {
    const u16* krow = kbase + (size_t)lm * OUT_;
#pragma unroll
    for (int c = 0; c < 4; ++c) B0[c] = ld32B(krow + c*32 + boff);
  }
  for (int m0 = 0; m0 < N_; m0 += 32) {
    const u16* krow1 = kbase + (size_t)(m0 + 16 + lm) * OUT_;
    __builtin_prefetch(krow1 + 32*OUT_, 0, 3);
#pragma unroll
    for (int c = 0; c < 4; ++c) B1[c] = ld32B(krow1 + c*32 + boff);   // tile m0+16
    v8f ca = vzero();
#pragma unroll
    for (int c = 0; c < 4; ++c) ca = WMMA_BF16(Aq[c], B0[c], ca);     // tile m0
#pragma unroll
    for (int r = 0; r < 8; ++r) sacc[r] += sigmoidf_(ca[r]);

    const int mn = (m0 + 32 < N_) ? (m0 + 32) : 0;                    // wrap-safe addr
    const u16* krow2 = kbase + (size_t)(mn + lm) * OUT_;
#pragma unroll
    for (int c = 0; c < 4; ++c) B0[c] = ld32B(krow2 + c*32 + boff);   // tile m0+32
    v8f cb = vzero();
#pragma unroll
    for (int c = 0; c < 4; ++c) cb = WMMA_BF16(Aq[c], B1[c], cb);     // tile m0+16
#pragma unroll
    for (int r = 0; r < 8; ++r) sacc[r] += sigmoidf_(cb[r]);
  }
#pragma unroll
  for (int off = 1; off < 16; off <<= 1) {
#pragma unroll
    for (int r = 0; r < 8; ++r) sacc[r] += __shfl_xor(sacc[r], off, 16);
  }
  if (lm == 0) {
    int jb = v*N_ + j0 + (hi ? 8 : 0);
#pragma unroll
    for (int r = 0; r < 8; ++r) rowsum[jb + r] = sacc[r];
  }
}

// ---------------------------------------------------------------------------
// VpT[v][d][j] = bf16( V[v][j][d] / rowsum[v][j] )
__global__ void k_vpt(const float* __restrict__ Vf, const float* __restrict__ rowsum,
                      u16* __restrict__ VpT) {
  int idx = blockIdx.x * blockDim.x + threadIdx.x;
  if (idx >= V_*OUT_*N_) return;
  int v   = idx >> 19;                   // OUT_*N_ = 2^19
  int rem = idx & ((OUT_*N_) - 1);
  int d   = rem >> 12;                   // N_ = 2^12
  int j   = rem & (N_ - 1);
  float x = Vf[((size_t)(v*N_ + j))*OUT_ + d] / rowsum[v*N_ + j];
  VpT[idx] = (u16)bf16_1(x);
}

// ---------------------------------------------------------------------------
// Fused output pass, 4 waves / block, async-LDS double buffering.
// out^T[d,i] = sum_j VpT[d,j] * sigmoid(kn_j . qn_i).
// Per j-chunk (32 j's): the block stages kb[32][128] and VpT[128][32] (16 KB)
// into LDS via GLOBAL_LOAD_ASYNC_TO_LDS_B128 while the previous chunk's 16
// WMMAs run; waves read operand fragments back with ds_load_b128.
__global__ __launch_bounds__(128)
void k_out(const u16* __restrict__ qb, const u16* __restrict__ kb,
           const u16* __restrict__ VpT, float* __restrict__ out)
{
  const int tid  = threadIdx.x;
  const int wave = tid >> 5;
  const int lane = tid & 31;
  const int lm   = lane & 15;
  const bool hi  = lane >= 16;

  const int blk = blockIdx.x;            // 192 blocks
  const int v   = blk / 64;
  const int i0  = (blk % 64) * 64 + wave * 16;

  __shared__ u16 skb[2][32*128];         // kb rows j0..j0+31   (8 KB each)
  __shared__ u16 svp[2][128*32];         // VpT rows, 32 j cols (8 KB each)

  // B operand of stage 1: qn i-tile as columns (per wave, global, hoisted)
  v16bf Bq[4];
  const u16* qrow = qb + ((size_t)(v*N_ + i0 + lm)) * OUT_;
#pragma unroll
  for (int c = 0; c < 4; ++c)
    Bq[c] = ld32B(qrow + c*32 + (hi ? 16 : 0));

  v8f D[8];
#pragma unroll
  for (int t = 0; t < 8; ++t) D[t] = vzero();

  const u16* kbase = kb  + (size_t)(v*N_)   * OUT_;
  const u16* pbase = VpT + (size_t)(v*OUT_) * N_;
  const int aoff = hi ? 16 : 0;

  // cooperative async staging of one j-chunk into LDS buffer `buf`
  auto stage = [&](int j0s, int buf) {
#pragma unroll
    for (int r = 0; r < 4; ++r) {                 // kb: 512 x 16B slots
      int flat = tid + r*128;
      int row = flat >> 4, c16 = flat & 15;
      async_g2l_b128(lds_off_of(&skb[buf][row*128 + c16*8]),
                     (u64)(uintptr_t)(kbase + (size_t)(j0s + row)*OUT_ + c16*8));
    }
#pragma unroll
    for (int r = 0; r < 4; ++r) {                 // VpT: 512 x 16B slots
      int flat = tid + r*128;
      int row = flat >> 2, c4 = flat & 3;
      async_g2l_b128(lds_off_of(&svp[buf][row*32 + c4*8]),
                     (u64)(uintptr_t)(pbase + (size_t)row*N_ + j0s + c4*8));
    }
  };

  stage(0, 0);
  wait_async0();
  __syncthreads();

  int buf = 0;
  for (int j0 = 0; j0 < N_; j0 += 32) {
    const int jn = (j0 + 32 < N_) ? (j0 + 32) : 0;      // wrap-safe last iter
    stage(jn, buf ^ 1);                                  // overlaps with compute

    const u16* skb_b = &skb[buf][0];
    const u16* svp_b = &svp[buf][0];

    // ---- stage 1: S^T tiles C1 (rows j0..j0+15) and C2 (rows j0+16..j0+31)
    v8f c1 = vzero(), c2 = vzero();
#pragma unroll
    for (int c = 0; c < 4; ++c) {
      int e = c*32 + aoff;
      v16bf A1 = ld16Bx2(skb_b + (size_t)lm*128 + e,       skb_b + (size_t)lm*128 + e + 8);
      v16bf A2 = ld16Bx2(skb_b + (size_t)(16+lm)*128 + e,  skb_b + (size_t)(16+lm)*128 + e + 8);
      c1 = WMMA_BF16(A1, Bq[c], c1);
      c2 = WMMA_BF16(A2, Bq[c], c2);
    }
#pragma unroll
    for (int r = 0; r < 8; ++r) { c1[r] = sigmoidf_(c1[r]); c2[r] = sigmoidf_(c2[r]); }

    // ---- C-layout -> bf16 B-operand layout (pack + half-swap)
    v8u bsv;
#pragma unroll
    for (int u = 0; u < 4; ++u) {
      u32 p1 = pkbf(c1[2*u], c1[2*u+1]);
      u32 p2 = pkbf(c2[2*u], c2[2*u+1]);
      u32 s1 = (u32)__shfl_xor((int)p1, 16, 32);
      u32 s2 = (u32)__shfl_xor((int)p2, 16, 32);
      bsv[u]     = hi ? s2 : p1;                  // K 0..7  | 16..23
      bsv[4 + u] = hi ? p2 : s1;                  // K 8..15 | 24..31
    }
    v16bf Bsv = __builtin_bit_cast(v16bf, bsv);

    // ---- stage 2: out^T[d-tile, i-tile] += VpT-tile (A) x S^T (B)
#pragma unroll
    for (int t = 0; t < 8; ++t) {
      const u16* prow = svp_b + (size_t)(t*16 + lm)*32 + aoff;
      v16bf Avp = ld16Bx2(prow, prow + 8);
      D[t] = WMMA_BF16(Avp, Bsv, D[t]);
    }

    wait_async0();        // next chunk's LDS data landed (loads ran under WMMAs)
    __syncthreads();      // all waves done reading `buf`, next stage may reuse it
    buf ^= 1;
  }

  // D[t]: M = d rows, N = i cols  ->  out[v][i][d]
  float* obase = out + (size_t)(v*N_) * OUT_;
#pragma unroll
  for (int t = 0; t < 8; ++t) {
#pragma unroll
    for (int r = 0; r < 8; ++r) {
      int d = t*16 + r + (hi ? 8 : 0);
      obase[(size_t)(i0 + lm) * OUT_ + d] = D[t][r];
    }
  }
}

// ---------------------------------------------------------------------------
extern "C" void kernel_launch(void* const* d_in, const int* in_sizes, int n_in,
                              void* d_out, int out_size, void* d_ws, size_t ws_size,
                              hipStream_t stream)
{
  (void)in_sizes; (void)n_in; (void)out_size; (void)ws_size;
  const float* X  = (const float*)d_in[0];
  const float* Wq = (const float*)d_in[1];
  const float* bq = (const float*)d_in[2];
  const float* Wk = (const float*)d_in[3];
  const float* bk = (const float*)d_in[4];
  const float* Wv = (const float*)d_in[5];
  const float* bv = (const float*)d_in[6];
  float* out = (float*)d_out;

  char* ws = (char*)d_ws;
  size_t off = 0;
  auto alloc = [&](size_t bytes) -> char* {
    char* p = ws + off;
    off += (bytes + 255) & ~(size_t)255;
    return p;
  };

  u16*   Xb     = (u16*)  alloc((size_t)V_*N_*IN_*2);
  u16*   Wqb    = (u16*)  alloc((size_t)V_*OUT_*IN_*2);
  u16*   Wkb    = (u16*)  alloc((size_t)V_*OUT_*IN_*2);
  u16*   Wvb    = (u16*)  alloc((size_t)V_*OUT_*IN_*2);
  float* Qf     = (float*)alloc((size_t)V_*N_*OUT_*4);
  float* Kf     = (float*)alloc((size_t)V_*N_*OUT_*4);
  float* Vf     = (float*)alloc((size_t)V_*N_*OUT_*4);
  float* part   = (float*)alloc((size_t)384*4);
  float* inv    = (float*)alloc(8*4);
  u16*   qbuf   = (u16*)  alloc((size_t)V_*N_*OUT_*2);
  u16*   kbuf   = (u16*)  alloc((size_t)V_*N_*OUT_*2);
  float* rowsum = (float*)alloc((size_t)V_*N_*4);
  u16*   VpT    = (u16*)  alloc((size_t)V_*OUT_*N_*2);

  const int nX = V_*N_*IN_;
  const int nW = V_*OUT_*IN_;
  const int nQ = V_*N_*OUT_;

  k_cvt_bf16<<<(nX+255)/256, 256, 0, stream>>>(X,  Xb,  nX);
  k_cvt_bf16<<<(nW+255)/256, 256, 0, stream>>>(Wq, Wqb, nW);
  k_cvt_bf16<<<(nW+255)/256, 256, 0, stream>>>(Wk, Wkb, nW);
  k_cvt_bf16<<<(nW+255)/256, 256, 0, stream>>>(Wv, Wvb, nW);

  k_proj<<<NWAVE_, 32, 0, stream>>>(Xb, Wqb, Wkb, Wvb, bq, bk, bv, Qf, Kf, Vf);
  k_sumsq<<<384, 256, 0, stream>>>(Qf, Kf, part);
  k_norm_reduce<<<1, 32, 0, stream>>>(part, inv);
  k_normalize<<<(nQ+255)/256, 256, 0, stream>>>(Qf, Kf, inv, qbuf, kbuf);
  k_rowsum<<<NWAVE_, 32, 0, stream>>>(qbuf, kbuf, rowsum);
  k_vpt<<<(nQ+255)/256, 256, 0, stream>>>(Vf, rowsum, VpT);
  k_out<<<192, 128, 0, stream>>>(qbuf, kbuf, VpT, out);
}
